// MultiHeadAttention_901943132652
// MI455X (gfx1250) — matrix operand order repaired and checked by reading, not compiled, not run on hardware
//
#include <hip/hip_runtime.h>

// -------- problem constants (match reference) --------
static constexpr int Bsz  = 4;
static constexpr int Slen = 2048;
static constexpr int Dm   = 1024;
static constexpr int Hn   = 16;
static constexpr int Dk   = 64;
#define LN_EPS 1e-5f

// -------- CDNA5 WMMA types --------
typedef __attribute__((ext_vector_type(16))) __bf16         v16bf;
typedef __attribute__((ext_vector_type(8)))  float          v8f;
typedef __attribute__((ext_vector_type(4)))  float          v4f;
typedef __attribute__((ext_vector_type(4)))  unsigned int   v4u;
typedef __attribute__((ext_vector_type(4)))  unsigned short v4us;

union FragBF {
  v4u            u[2];
  unsigned short s[16];
  v16bf          v;
};

__device__ __forceinline__ unsigned short f2bf(float f) {
  unsigned int x = __builtin_bit_cast(unsigned int, f);
  x += 0x7FFFu + ((x >> 16) & 1u);   // round-to-nearest-even
  return (unsigned short)(x >> 16);
}

// A-matrix 16x32 bf16 fragment: lane holds row M=lane&15; elem e -> K = 8*half + e (+16 for e>=8)
__device__ __forceinline__ v16bf load_frag_A(const unsigned short* __restrict__ p, int half) {
  FragBF f;
  f.u[0] = *(const v4u*)(p + 8 * half);
  f.u[1] = *(const v4u*)(p + 16 + 8 * half);
  return f.v;
}

// B-matrix 32x16 bf16 fragment: lane holds column N=lane&15; elem e -> K = 8*half + e (+16 for e>=8)
__device__ __forceinline__ v16bf load_frag_B(const unsigned short* __restrict__ p, int half) {
  FragBF f;
  f.u[0] = *(const v4u*)(p + 8 * half);
  f.u[1] = *(const v4u*)(p + 16 + 8 * half);
  return f.v;
}

// CDNA5 async DMA: copy 16B from LDS to global, tracked by ASYNCcnt (overlaps with compute).
__device__ __forceinline__ void async_b128_lds_to_global(void* g, void* l) {
  unsigned loff = (unsigned)(unsigned long long)l;  // low 32 bits of generic LDS ptr = LDS offset
  asm volatile("global_store_async_from_lds_b128 %0, %1, off th:TH_STORE_NT"
               :: "v"(g), "v"(loff)
               : "memory");
}

// -------- kernel 1: fp32 -> bf16 conversion (vectorized x4) --------
__global__ __launch_bounds__(256) void cvt_f32_bf16(const float* __restrict__ in,
                                                    unsigned short* __restrict__ out, int n4) {
  int i = blockIdx.x * 256 + threadIdx.x;
  if (i < n4) {
    v4f x = ((const v4f*)in)[i];
    v4us o;
#pragma unroll
    for (int e = 0; e < 4; ++e) o[e] = f2bf(x[e]);
    ((v4us*)out)[i] = o;
  }
}

// -------- kernel 2: P = X @ W^T + b (bf16 out). 4 waves/block, 4 M-tiles per wave (B reuse) ----
__global__ __launch_bounds__(128) void gemm_proj(const unsigned short* __restrict__ X,  // [M,K]
                                                 const unsigned short* __restrict__ W,  // [N,K]
                                                 const float* __restrict__ bias,        // [N]
                                                 unsigned short* __restrict__ Out,      // [M,N]
                                                 int M, int N, int K) {
  const int wave = threadIdx.x >> 5, lane = threadIdx.x & 31;
  const int half = lane >> 4, lrow = lane & 15;
  const int m0 = blockIdx.x * 64;
  const int n0 = (blockIdx.y * 4 + wave) * 16;

  const unsigned short* x0 = X + (size_t)(m0 + lrow) * K;
  const unsigned short* x1 = X + (size_t)(m0 + 16 + lrow) * K;
  const unsigned short* x2 = X + (size_t)(m0 + 32 + lrow) * K;
  const unsigned short* x3 = X + (size_t)(m0 + 48 + lrow) * K;
  const unsigned short* wr = W + (size_t)(n0 + lrow) * K;

  v8f c0 = {}, c1 = {}, c2 = {}, c3 = {};
  for (int k0 = 0; k0 < K; k0 += 32) {
    __builtin_prefetch(x0 + k0 + 128, 0, 1);
    v16bf b  = load_frag_B(wr + k0, half);
    v16bf a0 = load_frag_A(x0 + k0, half);
    v16bf a1 = load_frag_A(x1 + k0, half);
    v16bf a2 = load_frag_A(x2 + k0, half);
    v16bf a3 = load_frag_A(x3 + k0, half);
    c0 = __builtin_amdgcn_wmma_f32_16x16x32_bf16(false, a0, false, b, (short)0, c0, false, false);
    c1 = __builtin_amdgcn_wmma_f32_16x16x32_bf16(false, a1, false, b, (short)0, c1, false, false);
    c2 = __builtin_amdgcn_wmma_f32_16x16x32_bf16(false, a2, false, b, (short)0, c2, false, false);
    c3 = __builtin_amdgcn_wmma_f32_16x16x32_bf16(false, a3, false, b, (short)0, c3, false, false);
  }
  const int col = n0 + lrow;
  const float bval = bias[col];
  v8f acc[4] = {c0, c1, c2, c3};
#pragma unroll
  for (int t = 0; t < 4; ++t)
#pragma unroll
    for (int r = 0; r < 8; ++r) {
      int row = m0 + t * 16 + r + 8 * half;
      Out[(size_t)row * N + col] = f2bf(acc[t][r] + bval);
    }
}

// -------- kernel 3: per-head transpose Vp[B,S,Dm](head slice) -> Vt[B,H,Dk,S] --------
__global__ __launch_bounds__(256) void transpose_v(const unsigned short* __restrict__ Vp,
                                                   unsigned short* __restrict__ Vt) {
  __shared__ unsigned short t[64][65];
  const int s0 = blockIdx.x * 64, h = blockIdx.y, b = blockIdx.z;
  for (int i = threadIdx.x; i < 64 * 64; i += 256) {
    int r = i >> 6, c = i & 63;
    t[r][c] = Vp[(size_t)(b * Slen + s0 + r) * Dm + h * Dk + c];
  }
  __syncthreads();
  for (int i = threadIdx.x; i < 64 * 64; i += 256) {
    int d = i >> 6, c = i & 63;
    Vt[(((size_t)b * Hn + h) * Dk + d) * Slen + s0 + c] = t[c][d];
  }
}

// -------- kernel 4: attention core. 8 waves per (b,h,16-row q tile) --------
// dynamic LDS: sc = fp32 scores/probs [16][Slen] (128KB), sp = bf16 probs [16][Slen] (64KB)
__global__ __launch_bounds__(256) void attention(const unsigned short* __restrict__ Qp, // [B,S,Dm]
                                                 const unsigned short* __restrict__ Kp, // [B,S,Dm]
                                                 const unsigned short* __restrict__ Vt, // [B,H,Dk,S]
                                                 const int* __restrict__ mask,          // [B,1,S,S]
                                                 float* __restrict__ attn_out,          // [B,H,S,S]
                                                 unsigned short* __restrict__ ctx)      // [B,S,Dm]
{
  extern __shared__ unsigned char smem_dyn[];
  float*          sc = (float*)smem_dyn;                                  // 16*Slen fp32
  unsigned short* sp = (unsigned short*)(smem_dyn + 16 * Slen * 4);       // 16*Slen bf16

  const int qt = blockIdx.x, h = blockIdx.y, b = blockIdx.z;
  const int wave = threadIdx.x >> 5, lane = threadIdx.x & 31;
  const int half = lane >> 4, lrow = lane & 15;
  const size_t headoff = (size_t)h * Dk;

  // Q fragments (dk = 64 -> two K=32 fragments)
  const unsigned short* qrow = Qp + (size_t)(b * Slen + qt * 16 + lrow) * Dm + headoff;
  v16bf aq0 = load_frag_A(qrow, half);
  v16bf aq1 = load_frag_A(qrow + 32, half);

  // ---- phase 1: scores = Q K^T / sqrt(dk), masked, into LDS ----
  for (int j = wave; j < Slen / 16; j += 8) {
    const unsigned short* krow = Kp + (size_t)(b * Slen + j * 16 + lrow) * Dm + headoff;
    v16bf bk0 = load_frag_B(krow, half);
    v16bf bk1 = load_frag_B(krow + 32, half);
    v8f c = {};
    c = __builtin_amdgcn_wmma_f32_16x16x32_bf16(false, aq0, false, bk0, (short)0, c, false, false);
    c = __builtin_amdgcn_wmma_f32_16x16x32_bf16(false, aq1, false, bk1, (short)0, c, false, false);
    const int kcol = j * 16 + lrow;
#pragma unroll
    for (int r = 0; r < 8; ++r) {
      int qi = qt * 16 + r + 8 * half;
      float s = c[r] * 0.125f;  // 1/sqrt(64)
      if (mask[((size_t)b * Slen + qi) * Slen + kcol] == 0) s = -1e9f;
      sc[(r + 8 * half) * Slen + kcol] = s;
    }
  }
  __syncthreads();

  // ---- phase 2: softmax per row; fp32 probs -> LDS (async-DMA'd to HBM), bf16 probs -> LDS ----
#pragma unroll
  for (int rr = 0; rr < 2; ++rr) {
    const int row = wave + rr * 8;
    float*          srow  = sc + row * Slen;
    unsigned short* sprow = sp + row * Slen;
    float mx = -3.0e38f;
    for (int i = lane; i < Slen; i += 32) mx = fmaxf(mx, srow[i]);
    for (int o = 16; o > 0; o >>= 1) mx = fmaxf(mx, __shfl_xor(mx, o, 32));
    float sum = 0.f;
    for (int i = lane; i < Slen; i += 32) {
      float e = __expf(srow[i] - mx);
      srow[i] = e;
      sum += e;
    }
    for (int o = 16; o > 0; o >>= 1) sum += __shfl_xor(sum, o, 32);
    const float inv = 1.0f / sum;
    for (int i = lane; i < Slen; i += 32) {
      float p = srow[i] * inv;
      srow[i] = p;
      sprow[i] = f2bf(p);
    }
    // drain this wave's LDS stores, then stream the finished fp32 row to HBM via async DMA;
    // ASYNCcnt engine overlaps this 1GB-total writeback with the phase-3 WMMA loop below.
    asm volatile("s_wait_dscnt 0x0" ::: "memory");
    float* arow = attn_out + (((size_t)b * Hn + h) * Slen + (qt * 16 + row)) * Slen;
    for (int i = lane * 4; i < Slen; i += 128)
      async_b128_lds_to_global(arow + i, srow + i);
  }
  __syncthreads();

  // ---- phase 3: context = P @ Vh. waves 0..3 each own one 16x16 tile of [16 x 64] ----
  if (wave < 4) {
    const int d = wave * 16 + lrow;  // column within head
    const unsigned short* vtrow = Vt + (((size_t)b * Hn + h) * Dk + d) * Slen;
    const unsigned short* prow  = sp + lrow * Slen;
    v8f c = {};
    for (int k0 = 0; k0 < Slen; k0 += 32) {
      v16bf a  = load_frag_A(prow + k0, half);   // ds_load_b128 from LDS probs
      v16bf bb = load_frag_B(vtrow + k0, half);  // contiguous global_load_b128 (V transposed)
      c = __builtin_amdgcn_wmma_f32_16x16x32_bf16(false, a, false, bb, (short)0, c, false, false);
    }
#pragma unroll
    for (int r = 0; r < 8; ++r) {
      int srowi = qt * 16 + r + 8 * half;
      ctx[((size_t)b * Slen + srowi) * Dm + headoff + d] = f2bf(c[r]);
    }
  }
  // drain async attn writeback before the block's LDS can be reassigned
  asm volatile("s_wait_asynccnt 0x0" ::: "memory");
}

// -------- kernel 5: X = ctx @ Wo^T + bo + residual (fp32 out), same blocking as gemm_proj ----
__global__ __launch_bounds__(128) void gemm_out(const unsigned short* __restrict__ Cx, // [M,K]
                                                const unsigned short* __restrict__ W,  // [N,K]
                                                const float* __restrict__ bias,        // [N]
                                                const float* __restrict__ resid,       // [M,N]
                                                float* __restrict__ X,                 // [M,N]
                                                int M, int N, int K) {
  const int wave = threadIdx.x >> 5, lane = threadIdx.x & 31;
  const int half = lane >> 4, lrow = lane & 15;
  const int m0 = blockIdx.x * 64;
  const int n0 = (blockIdx.y * 4 + wave) * 16;

  const unsigned short* x0 = Cx + (size_t)(m0 + lrow) * K;
  const unsigned short* x1 = Cx + (size_t)(m0 + 16 + lrow) * K;
  const unsigned short* x2 = Cx + (size_t)(m0 + 32 + lrow) * K;
  const unsigned short* x3 = Cx + (size_t)(m0 + 48 + lrow) * K;
  const unsigned short* wr = W + (size_t)(n0 + lrow) * K;

  v8f c0 = {}, c1 = {}, c2 = {}, c3 = {};
  for (int k0 = 0; k0 < K; k0 += 32) {
    v16bf b  = load_frag_B(wr + k0, half);
    v16bf a0 = load_frag_A(x0 + k0, half);
    v16bf a1 = load_frag_A(x1 + k0, half);
    v16bf a2 = load_frag_A(x2 + k0, half);
    v16bf a3 = load_frag_A(x3 + k0, half);
    c0 = __builtin_amdgcn_wmma_f32_16x16x32_bf16(false, a0, false, b, (short)0, c0, false, false);
    c1 = __builtin_amdgcn_wmma_f32_16x16x32_bf16(false, a1, false, b, (short)0, c1, false, false);
    c2 = __builtin_amdgcn_wmma_f32_16x16x32_bf16(false, a2, false, b, (short)0, c2, false, false);
    c3 = __builtin_amdgcn_wmma_f32_16x16x32_bf16(false, a3, false, b, (short)0, c3, false, false);
  }
  const int col = n0 + lrow;
  const float bval = bias[col];
  v8f acc[4] = {c0, c1, c2, c3};
#pragma unroll
  for (int t = 0; t < 4; ++t)
#pragma unroll
    for (int r = 0; r < 8; ++r) {
      int row = m0 + t * 16 + r + 8 * half;
      X[(size_t)row * N + col] = acc[t][r] + bval + resid[(size_t)row * N + col];
    }
}

// -------- kernel 6: LayerNorm per row --------
__global__ __launch_bounds__(256) void layernorm(const float* __restrict__ X,
                                                 const float* __restrict__ gamma,
                                                 const float* __restrict__ beta,
                                                 float* __restrict__ Y, int N) {
  __shared__ float ssum[256], ssq[256];
  const int row = blockIdx.x, tid = threadIdx.x;
  const float* x = X + (size_t)row * N;
  float s = 0.f, s2 = 0.f;
  for (int i = tid; i < N; i += 256) { float v = x[i]; s += v; s2 += v * v; }
  ssum[tid] = s; ssq[tid] = s2;
  __syncthreads();
  for (int o = 128; o > 0; o >>= 1) {
    if (tid < o) { ssum[tid] += ssum[tid + o]; ssq[tid] += ssq[tid + o]; }
    __syncthreads();
  }
  const float mu  = ssum[0] / (float)N;
  const float var = ssq[0] / (float)N - mu * mu;
  const float inv = rsqrtf(var + LN_EPS);
  float* y = Y + (size_t)row * N;
  for (int i = tid; i < N; i += 256) y[i] = (x[i] - mu) * inv * gamma[i] + beta[i];
}

// -------- host orchestration --------
extern "C" void kernel_launch(void* const* d_in, const int* in_sizes, int n_in,
                              void* d_out, int out_size, void* d_ws, size_t ws_size,
                              hipStream_t stream) {
  (void)in_sizes; (void)n_in; (void)out_size; (void)ws_size;

  const float* q     = (const float*)d_in[0];
  const float* k     = (const float*)d_in[1];
  const float* v     = (const float*)d_in[2];
  const int*   mask  = (const int*)d_in[3];
  const float* wq    = (const float*)d_in[4];
  const float* bq    = (const float*)d_in[5];
  const float* wk    = (const float*)d_in[6];
  const float* bk    = (const float*)d_in[7];
  const float* wv    = (const float*)d_in[8];
  const float* bvv   = (const float*)d_in[9];
  const float* wo    = (const float*)d_in[10];
  const float* bo    = (const float*)d_in[11];
  const float* gamma = (const float*)d_in[12];
  const float* beta  = (const float*)d_in[13];

  const int    M  = Bsz * Slen;          // 8192 rows
  const size_t nx = (size_t)M * Dm;      // 8,388,608 activation elements
  const size_t nw = (size_t)Dm * Dm;     // 1,048,576 weight elements

  unsigned char* ws = (unsigned char*)d_ws;
  size_t off = 0;
  auto take = [&](size_t bytes) {
    unsigned char* p = ws + off;
    off += (bytes + 255) & ~(size_t)255;
    return p;
  };
  unsigned short* qbf = (unsigned short*)take(nx * 2);
  unsigned short* kbf = (unsigned short*)take(nx * 2);
  unsigned short* vbf = (unsigned short*)take(nx * 2);
  unsigned short* wqb = (unsigned short*)take(nw * 2);
  unsigned short* wkb = (unsigned short*)take(nw * 2);
  unsigned short* wvb = (unsigned short*)take(nw * 2);
  unsigned short* wob = (unsigned short*)take(nw * 2);
  unsigned short* Qp  = (unsigned short*)take(nx * 2);
  unsigned short* Kp  = (unsigned short*)take(nx * 2);
  unsigned short* Vp  = (unsigned short*)take(nx * 2);
  unsigned short* Vt  = (unsigned short*)take(nx * 2);
  unsigned short* ctx = (unsigned short*)take(nx * 2);
  float*          xln = (float*)take(nx * 4);

  float* y    = (float*)d_out;
  float* attn = y + nx;  // outputs concatenated: y [B,S,Dm] then attn [B,H,S,S]

  // 1) bf16 conversions
  cvt_f32_bf16<<<(unsigned)(nx / 4 + 255) / 256, 256, 0, stream>>>(q, qbf, (int)(nx / 4));
  cvt_f32_bf16<<<(unsigned)(nx / 4 + 255) / 256, 256, 0, stream>>>(k, kbf, (int)(nx / 4));
  cvt_f32_bf16<<<(unsigned)(nx / 4 + 255) / 256, 256, 0, stream>>>(v, vbf, (int)(nx / 4));
  cvt_f32_bf16<<<(unsigned)(nw / 4 + 255) / 256, 256, 0, stream>>>(wq, wqb, (int)(nw / 4));
  cvt_f32_bf16<<<(unsigned)(nw / 4 + 255) / 256, 256, 0, stream>>>(wk, wkb, (int)(nw / 4));
  cvt_f32_bf16<<<(unsigned)(nw / 4 + 255) / 256, 256, 0, stream>>>(wv, wvb, (int)(nw / 4));
  cvt_f32_bf16<<<(unsigned)(nw / 4 + 255) / 256, 256, 0, stream>>>(wo, wob, (int)(nw / 4));

  // 2) Q/K/V projections (WMMA, 64x64 block tiles)
  dim3 gg(M / 64, Dm / 64, 1);
  gemm_proj<<<gg, 128, 0, stream>>>(qbf, wqb, bq,  Qp, M, Dm, Dm);
  gemm_proj<<<gg, 128, 0, stream>>>(kbf, wkb, bk,  Kp, M, Dm, Dm);
  gemm_proj<<<gg, 128, 0, stream>>>(vbf, wvb, bvv, Vp, M, Dm, Dm);

  // 3) per-head V transpose for contiguous phase-3 B fragments
  dim3 gt(Slen / 64, Hn, Bsz);
  transpose_v<<<gt, 256, 0, stream>>>(Vp, Vt);

  // 4) attention (WMMA scores + softmax + async attn writeback + WMMA context)
  dim3 ga(Slen / 16, Hn, Bsz);
  const size_t attn_lds = (size_t)16 * Slen * 4 + (size_t)16 * Slen * 2;  // 192 KB
  attention<<<ga, 256, attn_lds, stream>>>(Qp, Kp, Vt, mask, attn, ctx);

  // 5) output projection + residual (WMMA)
  gemm_out<<<gg, 128, 0, stream>>>(ctx, wob, bo, q, xln, M, Dm, Dm);

  // 6) LayerNorm -> y
  layernorm<<<M, 256, 0, stream>>>(xln, gamma, beta, y, Dm);
}
